// Sequence_75651553951838
// MI455X (gfx1250) — matrix operand order repaired and checked by reading, not compiled, not run on hardware
//
#include <hip/hip_runtime.h>
#include <hip/hip_bf16.h>

typedef __attribute__((ext_vector_type(16))) _Float16 v16h;
typedef __attribute__((ext_vector_type(8)))  float    v8f;
typedef __attribute__((ext_vector_type(4)))  int      i4;    // 16B vector for LDS/global copies

#define HSZ 51
#define TT  1024
#define BB  1024
#define NGPP 256         // 4 gates x 64 padded columns
#define K1  64           // [h1(51) | x(1) | pad]
#define K2  128          // [h1(51)|pad | h2(51)|pad]
#define NWAVES 4         // one wave per SIMD32 of the WGP

// ---- byte offsets (identical in d_ws and LDS for the copied region) ----
#define OFF_W1   0                    // f16 [256][64]   = 32768 B
#define OFF_W2   32768                // f16 [256][128]  = 65536 B
#define OFF_B1   98304                // f32 [256]       = 1024 B
#define OFF_B2   99328                // f32 [256]       = 1024 B
#define OFF_WLIN 100352               // f32 [64]        = 256 B
#define OFF_BLIN 100608               // f32 [4]         = 16 B
#define COPY_BYTES 100624             // multiple of 16
// ---- LDS-only region: ping-pong h buffers ----
#define OFF_H1A  100624               // f16 [16][64]    = 2048 B
#define OFF_H1B  102672
#define OFF_H2A  104720
#define OFF_H2B  106768
#define SMEM_BYTES 108816

__device__ __forceinline__ float fast_sigmoid(float x) {
  float e = __builtin_amdgcn_exp2f(-1.44269504f * x);      // 1/(1+2^(-x*log2 e))
  return __builtin_amdgcn_rcpf(1.0f + e);
}
__device__ __forceinline__ float fast_tanh(float x) {
  // tanh(x) = 2/(1+2^(-2x*log2 e)) - 1, scale folded into the exp2 constant
  float e = __builtin_amdgcn_exp2f(-2.88539008f * x);
  return __builtin_fmaf(2.0f, __builtin_amdgcn_rcpf(1.0f + e), -1.0f);
}

union AF { v16h v; struct { i4 lo, hi; } q; };

// A fragment (16-bit A 16x32): lane<16 -> row=lane, K = kc+{0..7,16..23};
// lanes 16-31 -> row=lane-16, K = kc+{8..15,24..31}. Source: row-major f16 [16][64].
__device__ __forceinline__ v16h load_a(const _Float16* h, int kcElem, int ln, int lh) {
  const _Float16* p = h + ln * 64 + kcElem + lh * 8;
  AF f;
  f.q.lo = *(const i4*)(p);
  f.q.hi = *(const i4*)(p + 16);
  return f.v;
}

// B fragment (16-bit B 32x16): lane holds column N = nt*16 + (lane&15),
// K = kc*32 + (lane<16 ? {0..7,16..23} : {8..15,24..31}), two f16 per dword ascending.
__device__ __forceinline__ v16h load_b(const _Float16* w, int kpad, int nt, int kc, int ln, int lh) {
  const _Float16* p = w + (nt * 16 + ln) * kpad + kc * 32 + lh * 8;
  AF f;
  f.q.lo = *(const i4*)(p);
  f.q.hi = *(const i4*)(p + 16);
  return f.v;
}

// ------- weight repack: f32 -> gate-padded f16 (gate g at columns 64g..64g+50) -------
__global__ void lstm_prep(const float* __restrict__ Wih1, const float* __restrict__ Whh1,
                          const float* __restrict__ bih1, const float* __restrict__ bhh1,
                          const float* __restrict__ Wih2, const float* __restrict__ Whh2,
                          const float* __restrict__ bih2, const float* __restrict__ bhh2,
                          const float* __restrict__ Wlin, const float* __restrict__ blin,
                          char* __restrict__ ws) {
  _Float16* W1 = (_Float16*)(ws + OFF_W1);
  _Float16* W2 = (_Float16*)(ws + OFF_W2);
  float* B1 = (float*)(ws + OFF_B1);
  float* B2 = (float*)(ws + OFF_B2);
  float* WL = (float*)(ws + OFF_WLIN);
  float* BL = (float*)(ws + OFF_BLIN);
  int tid = threadIdx.x + blockIdx.x * blockDim.x;
  int nth = blockDim.x * gridDim.x;

  for (int i = tid; i < NGPP * K1; i += nth) {     // B1 row k: [W_hh1 | W_ih1(k==51) | 0]
    int n = i / K1, k = i % K1;
    int g = n >> 6, j = n & 63;
    float v = 0.f;
    if (j < HSZ) {
      int r = g * HSZ + j;                         // original gate row (PyTorch i,f,g,o order)
      if (k < HSZ) v = Whh1[r * HSZ + k];
      else if (k == HSZ) v = Wih1[r];
    }
    W1[i] = (_Float16)v;
  }
  for (int i = tid; i < NGPP * K2; i += nth) {     // B2 rows: [W_ih2 | 0 | W_hh2 | 0]
    int n = i / K2, k = i % K2;
    int g = n >> 6, j = n & 63;
    float v = 0.f;
    if (j < HSZ) {
      int r = g * HSZ + j;
      if (k < HSZ) v = Wih2[r * HSZ + k];
      else if (k >= 64 && k < 64 + HSZ) v = Whh2[r * HSZ + (k - 64)];
    } else if (n == HSZ && k >= 64 && k < 64 + HSZ) {
      v = Wlin[k - 64];                            // hide W_lin in dead column 51: y = h2_old . W_lin
    }
    W2[i] = (_Float16)v;
  }
  for (int n = tid; n < NGPP; n += nth) {
    int g = n >> 6, j = n & 63;
    float b1 = 0.f, b2 = 0.f;
    if (j < HSZ) {
      int r = g * HSZ + j;
      b1 = bih1[r] + bhh1[r];
      b2 = bih2[r] + bhh2[r];
    }
    B1[n] = b1;
    B2[n] = b2;                                    // column 51 bias stays 0 (y column)
  }
  for (int k = tid; k < K1; k += nth) WL[k] = (k < HSZ) ? Wlin[k] : 0.f;
  if (tid == 0) { BL[0] = blin[0]; BL[1] = 0.f; BL[2] = 0.f; BL[3] = 0.f; }
}

// ---- persistent LSTM: 1 block = 16 batch rows; wave w owns gate-column j = 16w+ln for ALL
// four gates (tiles 4g+w) -> c/h update is register-resident; layer-2 weights live in VGPRs ----
__global__ __launch_bounds__(32 * NWAVES) void lstm_persistent(const float* __restrict__ in,
                                                               const char* __restrict__ ws,
                                                               float* __restrict__ out) {
  extern __shared__ char smem[];
  const int tid  = threadIdx.x;        // 0..127
  const int l32  = tid & 31;
  const int w    = tid >> 5;           // wave id 0..3
  const int ln   = l32 & 15;
  const int lh   = (l32 >> 4) & 1;
  const int mb   = (l32 < 16) ? 0 : 8; // C/D fragment row base
  const int row0 = blockIdx.x * 16;
  const int j    = 16 * w + ln;        // hidden column owned by this lane (valid if < 51)
  const bool jok = (j < HSZ);
  const bool yext = (w == 3) && (ln == 3);  // lane holding the hidden y column (n=51, gate 0)

  // Stage packed weights/biases from global scratch into LDS.
  {
    const i4* src = (const i4*)ws;
    i4* dst = (i4*)smem;
    for (int i = tid; i < COPY_BYTES / 16; i += 32 * NWAVES) dst[i] = src[i];
  }
  // Zero all four h ping-pong buffers (incl. x-slot and K-padding).
  {
    float* z = (float*)(smem + OFF_H1A);
    for (int i = tid; i < (SMEM_BYTES - OFF_H1A) / 4; i += 32 * NWAVES) z[i] = 0.f;
  }
  __syncthreads();

  _Float16* W1 = (_Float16*)(smem + OFF_W1);
  _Float16* W2 = (_Float16*)(smem + OFF_W2);
  const float* Bi1 = (const float*)(smem + OFF_B1);
  const float* Bi2 = (const float*)(smem + OFF_B2);
  const float* WL  = (const float*)(smem + OFF_WLIN);
  const float  bl  = *(const float*)(smem + OFF_BLIN);
  _Float16* h1r = (_Float16*)(smem + OFF_H1A);
  _Float16* h1w = (_Float16*)(smem + OFF_H1B);
  _Float16* h2r = (_Float16*)(smem + OFF_H2A);
  _Float16* h2w = (_Float16*)(smem + OFF_H2B);
  const float* inRow = in + (size_t)(row0 + tid) * TT;   // deref'd only when tid<16
  float* outRow = out + (size_t)(row0 + tid) * TT;

  // Layer-2 B fragments: resident in VGPRs for the whole sequence (16 x v16h = 128 VGPRs).
  v16h B2f[4][4];
#pragma unroll
  for (int g = 0; g < 4; ++g)
#pragma unroll
    for (int kc = 0; kc < 4; ++kc) B2f[g][kc] = load_b(W2, K2, 4 * g + w, kc, ln, lh);

  // Per-lane gate biases (i,f,g,o at n = 64g + j).
  float bs1[4], bs2[4];
#pragma unroll
  for (int g = 0; g < 4; ++g) { bs1[g] = Bi1[64 * g + j]; bs2[g] = Bi2[64 * g + j]; }

  // Register-resident cell state: 8 rows per lane, column j.
  float c1[8] = {}, c2[8] = {};

  // Prologue: inject x_0 into the read buffer's augmented-K slot (K=51).
  if (tid < 16) h1r[tid * 64 + HSZ] = (_Float16)inRow[0];
  __syncthreads();

  for (int t = 0; t < TT; ++t) {
    // Next-step input, issued early (consumed at end of layer-1 phase).
    float xnext = 0.f;
    if (tid < 16) xnext = inRow[(t + 1 < TT) ? (t + 1) : (TT - 1)];

    // h2_{t-1} A-fragments: h2r is stable for the whole iteration, so issue these
    // before layer 1 — their latency hides behind the entire layer-1 phase.
    v16h a22 = load_a(h2r, 0,  ln, lh);
    v16h a23 = load_a(h2r, 32, ln, lh);

    // ================= Layer 1: gates = [h1|x] x B1; update c1/h1 in registers =================
    {
      v16h a0 = load_a(h1r, 0,  ln, lh);
      v16h a1 = load_a(h1r, 32, ln, lh);
      v8f ai = {}, af = {}, ag = {}, ao = {};
      ai = __builtin_amdgcn_wmma_f32_16x16x32_f16(false, a0, false, load_b(W1, K1, w,      0, ln, lh), (short)0, ai, false, false);
      ai = __builtin_amdgcn_wmma_f32_16x16x32_f16(false, a1, false, load_b(W1, K1, w,      1, ln, lh), (short)0, ai, false, false);
      af = __builtin_amdgcn_wmma_f32_16x16x32_f16(false, a0, false, load_b(W1, K1, 4 + w,  0, ln, lh), (short)0, af, false, false);
      af = __builtin_amdgcn_wmma_f32_16x16x32_f16(false, a1, false, load_b(W1, K1, 4 + w,  1, ln, lh), (short)0, af, false, false);
      ag = __builtin_amdgcn_wmma_f32_16x16x32_f16(false, a0, false, load_b(W1, K1, 8 + w,  0, ln, lh), (short)0, ag, false, false);
      ag = __builtin_amdgcn_wmma_f32_16x16x32_f16(false, a1, false, load_b(W1, K1, 8 + w,  1, ln, lh), (short)0, ag, false, false);
      ao = __builtin_amdgcn_wmma_f32_16x16x32_f16(false, a0, false, load_b(W1, K1, 12 + w, 0, ln, lh), (short)0, ao, false, false);
      ao = __builtin_amdgcn_wmma_f32_16x16x32_f16(false, a1, false, load_b(W1, K1, 12 + w, 1, ln, lh), (short)0, ao, false, false);
#pragma unroll
      for (int r = 0; r < 8; ++r) {
        float iv = fast_sigmoid(ai[r] + bs1[0]);
        float fv = fast_sigmoid(af[r] + bs1[1]);
        float gv = fast_tanh   (ag[r] + bs1[2]);
        float ov = fast_sigmoid(ao[r] + bs1[3]);
        float c  = __builtin_fmaf(fv, c1[r], iv * gv);
        c1[r] = c;
        float h = ov * fast_tanh(c);
        if (jok) h1w[(mb + r) * 64 + j] = (_Float16)h;
      }
      if (tid < 16) h1w[tid * 64 + HSZ] = (_Float16)xnext;   // x_{t+1} for next step
    }
    __syncthreads();                                         // h1_new (+x) visible

    // ===== Layer 2: gates = [h1new | h2] x B2 (register-resident weights); y folded in =====
    {
      v16h a20 = load_a(h1w, 0,  ln, lh);
      v16h a21 = load_a(h1w, 32, ln, lh);
      v8f ai = {}, af = {}, ag = {}, ao = {};
      ai = __builtin_amdgcn_wmma_f32_16x16x32_f16(false, a20, false, B2f[0][0], (short)0, ai, false, false);
      ai = __builtin_amdgcn_wmma_f32_16x16x32_f16(false, a21, false, B2f[0][1], (short)0, ai, false, false);
      ai = __builtin_amdgcn_wmma_f32_16x16x32_f16(false, a22, false, B2f[0][2], (short)0, ai, false, false);
      ai = __builtin_amdgcn_wmma_f32_16x16x32_f16(false, a23, false, B2f[0][3], (short)0, ai, false, false);
      af = __builtin_amdgcn_wmma_f32_16x16x32_f16(false, a20, false, B2f[1][0], (short)0, af, false, false);
      af = __builtin_amdgcn_wmma_f32_16x16x32_f16(false, a21, false, B2f[1][1], (short)0, af, false, false);
      af = __builtin_amdgcn_wmma_f32_16x16x32_f16(false, a22, false, B2f[1][2], (short)0, af, false, false);
      af = __builtin_amdgcn_wmma_f32_16x16x32_f16(false, a23, false, B2f[1][3], (short)0, af, false, false);
      ag = __builtin_amdgcn_wmma_f32_16x16x32_f16(false, a20, false, B2f[2][0], (short)0, ag, false, false);
      ag = __builtin_amdgcn_wmma_f32_16x16x32_f16(false, a21, false, B2f[2][1], (short)0, ag, false, false);
      ag = __builtin_amdgcn_wmma_f32_16x16x32_f16(false, a22, false, B2f[2][2], (short)0, ag, false, false);
      ag = __builtin_amdgcn_wmma_f32_16x16x32_f16(false, a23, false, B2f[2][3], (short)0, ag, false, false);
      ao = __builtin_amdgcn_wmma_f32_16x16x32_f16(false, a20, false, B2f[3][0], (short)0, ao, false, false);
      ao = __builtin_amdgcn_wmma_f32_16x16x32_f16(false, a21, false, B2f[3][1], (short)0, ao, false, false);
      ao = __builtin_amdgcn_wmma_f32_16x16x32_f16(false, a22, false, B2f[3][2], (short)0, ao, false, false);
      ao = __builtin_amdgcn_wmma_f32_16x16x32_f16(false, a23, false, B2f[3][3], (short)0, ao, false, false);

      // Hidden y column (n=51): gate-0 acc of lane {w=3, ln=3} is y_{t-1} - b_lin (bias2[51]=0).
      if (yext && t > 0) {
#pragma unroll
        for (int r = 0; r < 8; ++r) out[(size_t)(row0 + mb + r) * TT + (t - 1)] = ai[r] + bl;
      }
#pragma unroll
      for (int r = 0; r < 8; ++r) {
        float iv = fast_sigmoid(ai[r] + bs2[0]);
        float fv = fast_sigmoid(af[r] + bs2[1]);
        float gv = fast_tanh   (ag[r] + bs2[2]);
        float ov = fast_sigmoid(ao[r] + bs2[3]);
        float c  = __builtin_fmaf(fv, c2[r], iv * gv);
        c2[r] = c;
        float h = ov * fast_tanh(c);
        if (jok) h2w[(mb + r) * 64 + j] = (_Float16)h;
      }
    }
    __syncthreads();                                         // h2_new visible

    // swap ping-pong buffers
    _Float16* tp;
    tp = h1r; h1r = h1w; h1w = tp;
    tp = h2r; h2r = h2w; h2w = tp;
  }

  // Final output y_{T-1} = h2_{T-1} . W_lin + b_lin (h2 last written is now h2r).
  if (tid < 16) {
    float y = bl;
    for (int k = 0; k < HSZ; ++k) y += (float)h2r[tid * 64 + k] * WL[k];
    outRow[TT - 1] = y;
  }
}

extern "C" void kernel_launch(void* const* d_in, const int* in_sizes, int n_in,
                              void* d_out, int out_size, void* d_ws, size_t ws_size,
                              hipStream_t stream) {
  (void)in_sizes; (void)n_in; (void)out_size; (void)ws_size;
  const float* in_  = (const float*)d_in[0];
  const float* Wih1 = (const float*)d_in[1];
  const float* Whh1 = (const float*)d_in[2];
  const float* bih1 = (const float*)d_in[3];
  const float* bhh1 = (const float*)d_in[4];
  const float* Wih2 = (const float*)d_in[5];
  const float* Whh2 = (const float*)d_in[6];
  const float* bih2 = (const float*)d_in[7];
  const float* bhh2 = (const float*)d_in[8];
  const float* Wlin = (const float*)d_in[9];
  const float* blin = (const float*)d_in[10];

  (void)hipFuncSetAttribute((const void*)lstm_persistent,
                            hipFuncAttributeMaxDynamicSharedMemorySize, SMEM_BYTES);

  lstm_prep<<<dim3(4), dim3(256), 0, stream>>>(Wih1, Whh1, bih1, bhh1,
                                               Wih2, Whh2, bih2, bhh2,
                                               Wlin, blin, (char*)d_ws);
  lstm_persistent<<<dim3(BB / 16), dim3(32 * NWAVES), SMEM_BYTES, stream>>>(
      in_, (const char*)d_ws, (float*)d_out);
}
